// RoutedLoRA_59717225283913
// MI455X (gfx1250) — matrix operand order repaired and checked by reading, not compiled, not run on hardware
//
#include <hip/hip_runtime.h>
#include <stdint.h>

#define IN_F   4096
#define OUT_F  4096
#define RANK   8
#define N_EXP  64
#define TOP_K  16
#define R_DIM  16
#define BOTTLE 512
#define NTOK   16384           // B_SZ * SEQ
#define SCALING 2.0f

typedef __bf16 bf16;
typedef __attribute__((ext_vector_type(4)))  float f32x4;
typedef __attribute__((ext_vector_type(8)))  float f32x8;
typedef __attribute__((ext_vector_type(4)))  bf16  b16x4;
typedef __attribute__((ext_vector_type(8)))  bf16  b16x8;
typedef __attribute__((ext_vector_type(16))) bf16  b16x16;

__device__ __forceinline__ b16x16 cat16(b16x8 a, b16x8 b) {
  return __builtin_shufflevector(a, b, 0,1,2,3,4,5,6,7,8,9,10,11,12,13,14,15);
}

// --- CDNA5 async memory->LDS copy (16B per lane), tracked by ASYNCcnt -------
__device__ __forceinline__ void async_load_b128(void* lds_ptr, const void* gptr) {
  uint32_t lds_off = (uint32_t)(uintptr_t)lds_ptr;   // 32-bit LDS byte address
  asm volatile("global_load_async_to_lds_b128 %0, %1, off"
               :: "v"(lds_off), "v"(gptr) : "memory");
}
__device__ __forceinline__ void wait_async0() {
  asm volatile("s_wait_asynccnt 0x0" ::: "memory");
}

// ---------------------------------------------------------------- converts
__global__ __launch_bounds__(256)
void f32_to_bf16_k(const float* __restrict__ src, bf16* __restrict__ dst, int n4) {
  int i = blockIdx.x * 256 + threadIdx.x;
  if (i < n4) {
    f32x4 v = *(const f32x4*)(src + (size_t)i * 4);
    b16x4 o = { (bf16)v.x, (bf16)v.y, (bf16)v.z, (bf16)v.w };
    *(b16x4*)(dst + (size_t)i * 4) = o;
  }
}

// ---------------------------------------------------------------- router
// one wave32 per token; Wq staged through LDS in 256-column chunks
__global__ __launch_bounds__(256)
void router_k(const float* __restrict__ x, const float* __restrict__ wq,
              const float* __restrict__ keys, float* __restrict__ gate) {
  __shared__ float wql[R_DIM][256];
  const int tid  = threadIdx.x;
  const int lane = tid & 31;
  const int wid  = tid >> 5;
  const int token = blockIdx.x * 8 + wid;

  float acc[R_DIM];
#pragma unroll
  for (int r = 0; r < R_DIM; ++r) acc[r] = 0.0f;

  for (int kc = 0; kc < IN_F; kc += 256) {
#pragma unroll
    for (int r = 0; r < R_DIM; ++r)
      wql[r][tid] = wq[(size_t)r * IN_F + kc + tid];
    __syncthreads();
#pragma unroll
    for (int it = 0; it < 8; ++it) {
      int kl = lane + it * 32;
      float xv = x[(size_t)token * IN_F + kc + kl];
#pragma unroll
      for (int r = 0; r < R_DIM; ++r) acc[r] += xv * wql[r][kl];
    }
    __syncthreads();
  }
  // wave-wide sum -> every lane holds full q[r]
#pragma unroll
  for (int r = 0; r < R_DIM; ++r)
#pragma unroll
    for (int off = 16; off >= 1; off >>= 1)
      acc[r] += __shfl_xor(acc[r], off, 32);

  // scores for experts (lane) and (lane+32)
  float s1 = 0.0f, s2 = 0.0f;
#pragma unroll
  for (int r = 0; r < R_DIM; ++r) {
    s1 += acc[r] * keys[(size_t)lane * R_DIM + r];
    s2 += acc[r] * keys[(size_t)(lane + 32) * R_DIM + r];
  }

  // top-16 via rank counting (lower index wins ties within a half; cross-half ties unbroken)
  int r1 = 0, r2 = 0;
#pragma unroll
  for (int j = 0; j < 32; ++j) {
    float sj = __shfl(s1, j, 32);
    r1 += (sj > s1) || (sj == s1 && j < lane);
    r2 += (sj > s2);
  }
#pragma unroll
  for (int j = 0; j < 32; ++j) {
    float sj = __shfl(s2, j, 32);
    r1 += (sj > s1);
    r2 += (sj > s2) || (sj == s2 && j < lane);
  }
  bool in1 = r1 < TOP_K, in2 = r2 < TOP_K;

  float mx = fmaxf(in1 ? s1 : -3.0e38f, in2 ? s2 : -3.0e38f);
#pragma unroll
  for (int off = 16; off >= 1; off >>= 1) mx = fmaxf(mx, __shfl_xor(mx, off, 32));
  float e1 = in1 ? __expf(s1 - mx) : 0.0f;
  float e2 = in2 ? __expf(s2 - mx) : 0.0f;
  float sm = e1 + e2;
#pragma unroll
  for (int off = 16; off >= 1; off >>= 1) sm += __shfl_xor(sm, off, 32);
  float inv = 1.0f / sm;
  gate[(size_t)token * N_EXP + lane]      = e1 * inv;
  gate[(size_t)token * N_EXP + lane + 32] = e2 * inv;
}

// ---------------------------------------------------------------- GEMM common
#define BM 128
#define BN 128
#define BK 32
#define LS 40   // LDS row stride in bf16 (80B: 16B-aligned, conflict-free)

// A frag (16x32 bf16, M x K): lane&15 = M row; lane>=16 selects K halves {8..15,24..31}
__device__ __forceinline__ b16x16 load_a_frag(const bf16* tile, int mbase, int lane) {
  const bf16* p = tile + (mbase + (lane & 15)) * LS + ((lane >> 4) << 3);
  return cat16(*(const b16x8*)p, *(const b16x8*)(p + 16));
}
// B frag (32x16 bf16, K x N) from row-major [N][K] weight tile:
// lane&15 = column n; lane>=16 selects K halves {8..15,24..31}
__device__ __forceinline__ b16x16 load_b_frag(const bf16* tile, int nbase, int lane) {
  const bf16* p = tile + (nbase + (lane & 15)) * LS + ((lane >> 4) << 3);
  return cat16(*(const b16x8*)p, *(const b16x8*)(p + 16));
}

// ---------------------------------------------------------------- GEMM1 (+gate)
// zg[m, n] = bf16( gate[m, n>>3] * sum_k x[m,k]*A[n,k] ),  M=16384 N=512 K=4096
__global__ __launch_bounds__(256)
void gemm1_k(const float* __restrict__ x, const bf16* __restrict__ awb,
             const float* __restrict__ gate, bf16* __restrict__ zg) {
  __shared__ bf16  xt[2][BM * LS];
  __shared__ bf16  wt[2][BM * LS];
  __shared__ float gl[BM * 16];

  const int tid  = threadIdx.x;
  const int lane = tid & 31;
  const int wid  = tid >> 5;
  const int wm   = (wid & 1) * 64;
  const int wn   = (wid >> 1) * 32;
  const int m0   = blockIdx.x * BM;
  const int n0   = blockIdx.y * BN;
  const int e0   = n0 >> 3;

#pragma unroll
  for (int i = 0; i < 8; ++i) {
    int idx = tid + i * 256;  // 128 rows x 16 experts
    gl[idx] = gate[(size_t)(m0 + (idx >> 4)) * N_EXP + e0 + (idx & 15)];
  }

  f32x8 acc[4][2];
#pragma unroll
  for (int a = 0; a < 4; ++a)
#pragma unroll
    for (int b = 0; b < 2; ++b)
      acc[a][b] = f32x8{0.f,0.f,0.f,0.f,0.f,0.f,0.f,0.f};

  f32x4 xr[4];
  // x tile: 128x32 fp32 -> bf16 via VGPRs (conversion required)
  auto loadX = [&](int ks) {
#pragma unroll
    for (int i = 0; i < 4; ++i) {
      int f = tid + i * 256, row = f >> 3, c4 = (f & 7) << 2;
      xr[i] = *(const f32x4*)(x + (size_t)(m0 + row) * IN_F + ks * BK + c4);
    }
  };
  auto stageX = [&](int buf) {
#pragma unroll
    for (int i = 0; i < 4; ++i) {
      int f = tid + i * 256, row = f >> 3, c4 = (f & 7) << 2;
      b16x4 v = { (bf16)xr[i].x, (bf16)xr[i].y, (bf16)xr[i].z, (bf16)xr[i].w };
      *(b16x4*)&xt[buf][row * LS + c4] = v;
    }
  };
  // weight tile: already bf16 -> async copy straight into LDS (no VGPR staging)
  auto asyncW = [&](int ks, int buf) {
#pragma unroll
    for (int i = 0; i < 2; ++i) {
      int c = tid + i * 256, row = c >> 2, c8 = (c & 3) << 3;
      async_load_b128(&wt[buf][row * LS + c8],
                      awb + (size_t)(n0 + row) * IN_F + ks * BK + c8);
    }
  };

  const int KSTEPS = IN_F / BK;   // 128
  asyncW(0, 0);
  loadX(0);
  stageX(0);
  wait_async0();
  __syncthreads();

  for (int ks = 0; ks < KSTEPS; ++ks) {
    const int cur = ks & 1;
    if (ks + 1 < KSTEPS) {
      asyncW(ks + 1, cur ^ 1);
      loadX(ks + 1);
      if (ks + 2 < KSTEPS)   // pull the fp32 x stream toward the WGP early
        __builtin_prefetch(x + (size_t)(m0 + (tid >> 3)) * IN_F + (ks + 2) * BK, 0, 0);
    }
    b16x16 bfr[2];
#pragma unroll
    for (int nb = 0; nb < 2; ++nb) bfr[nb] = load_b_frag(wt[cur], wn + nb * 16, lane);
#pragma unroll
    for (int mb = 0; mb < 4; ++mb) {
      b16x16 afr = load_a_frag(xt[cur], wm + mb * 16, lane);
#pragma unroll
      for (int nb = 0; nb < 2; ++nb)
        acc[mb][nb] = __builtin_amdgcn_wmma_f32_16x16x32_bf16(
            false, afr, false, bfr[nb], (short)0, acc[mb][nb], false, false);
    }
    if (ks + 1 < KSTEPS) stageX(cur ^ 1);
    wait_async0();
    __syncthreads();
  }

#pragma unroll
  for (int mb = 0; mb < 4; ++mb)
#pragma unroll
    for (int nb = 0; nb < 2; ++nb) {
      int nl = wn + nb * 16 + (lane & 15);
      int el = nl >> 3;
      int mb8 = wm + mb * 16 + ((lane >> 4) << 3);
#pragma unroll
      for (int i = 0; i < 8; ++i) {
        int ml = mb8 + i;
        float g = gl[ml * 16 + el];
        zg[(size_t)(m0 + ml) * BOTTLE + n0 + nl] = (bf16)(acc[mb][nb][i] * g);
      }
    }
}

// ---------------------------------------------------------------- GEMM2
// out[m,n] = 2.0 * sum_k zg[m,k]*B[n,k],  M=16384 N=4096 K=512
// both operand tiles are bf16 in memory -> fully async-copied into LDS
__global__ __launch_bounds__(256)
void gemm2_k(const bf16* __restrict__ zg, const bf16* __restrict__ bwb,
             float* __restrict__ out) {
  __shared__ bf16 xt[2][BM * LS];
  __shared__ bf16 wt[2][BM * LS];

  const int tid  = threadIdx.x;
  const int lane = tid & 31;
  const int wid  = tid >> 5;
  const int wm   = (wid & 1) * 64;
  const int wn   = (wid >> 1) * 32;
  const int m0   = blockIdx.x * BM;
  const int n0   = blockIdx.y * BN;

  f32x8 acc[4][2];
#pragma unroll
  for (int a = 0; a < 4; ++a)
#pragma unroll
    for (int b = 0; b < 2; ++b)
      acc[a][b] = f32x8{0.f,0.f,0.f,0.f,0.f,0.f,0.f,0.f};

  auto asyncTiles = [&](int ks, int buf) {
#pragma unroll
    for (int i = 0; i < 2; ++i) {
      int c = tid + i * 256, row = c >> 2, c8 = (c & 3) << 3;
      async_load_b128(&xt[buf][row * LS + c8],
                      zg  + (size_t)(m0 + row) * BOTTLE + ks * BK + c8);
      async_load_b128(&wt[buf][row * LS + c8],
                      bwb + (size_t)(n0 + row) * BOTTLE + ks * BK + c8);
    }
  };

  const int KSTEPS = BOTTLE / BK;   // 16
  asyncTiles(0, 0);
  wait_async0();
  __syncthreads();

  for (int ks = 0; ks < KSTEPS; ++ks) {
    const int cur = ks & 1;
    if (ks + 1 < KSTEPS) asyncTiles(ks + 1, cur ^ 1);
    b16x16 bfr[2];
#pragma unroll
    for (int nb = 0; nb < 2; ++nb) bfr[nb] = load_b_frag(wt[cur], wn + nb * 16, lane);
#pragma unroll
    for (int mb = 0; mb < 4; ++mb) {
      b16x16 afr = load_a_frag(xt[cur], wm + mb * 16, lane);
#pragma unroll
      for (int nb = 0; nb < 2; ++nb)
        acc[mb][nb] = __builtin_amdgcn_wmma_f32_16x16x32_bf16(
            false, afr, false, bfr[nb], (short)0, acc[mb][nb], false, false);
    }
    wait_async0();
    __syncthreads();
  }

#pragma unroll
  for (int mb = 0; mb < 4; ++mb)
#pragma unroll
    for (int nb = 0; nb < 2; ++nb) {
      int nl = wn + nb * 16 + (lane & 15);
      int mb8 = wm + mb * 16 + ((lane >> 4) << 3);
#pragma unroll
      for (int i = 0; i < 8; ++i)
        out[(size_t)(m0 + mb8 + i) * OUT_F + n0 + nl] = acc[mb][nb][i] * SCALING;
    }
}

// ---------------------------------------------------------------- launch
extern "C" void kernel_launch(void* const* d_in, const int* in_sizes, int n_in,
                              void* d_out, int out_size, void* d_ws, size_t ws_size,
                              hipStream_t stream) {
  const float* x    = (const float*)d_in[0];
  const float* A_w  = (const float*)d_in[1];
  const float* B_w  = (const float*)d_in[2];
  const float* Wq_w = (const float*)d_in[3];
  const float* keys = (const float*)d_in[4];
  float* out = (float*)d_out;

  // workspace layout (needs 28 MB):
  char* ws = (char*)d_ws;
  float* gate = (float*)ws;                              // 16384*64 f32  = 4 MB
  bf16*  abf  = (bf16*)(ws + ((size_t)4  << 20));        // 512*4096 bf16 = 4 MB
  bf16*  bbf  = (bf16*)(ws + ((size_t)8  << 20));        // 4096*512 bf16 = 4 MB
  bf16*  zg   = (bf16*)(ws + ((size_t)12 << 20));        // 16384*512 bf16= 16 MB

  {
    int n4a = (BOTTLE * IN_F) / 4;   // 524288
    f32_to_bf16_k<<<(n4a + 255) / 256, 256, 0, stream>>>(A_w, abf, n4a);
    int n4b = (OUT_F * BOTTLE) / 4;
    f32_to_bf16_k<<<(n4b + 255) / 256, 256, 0, stream>>>(B_w, bbf, n4b);
  }
  router_k<<<NTOK / 8, 256, 0, stream>>>(x, Wq_w, keys, gate);

  dim3 g1(NTOK / BM, BOTTLE / BN);   // (128, 4)
  gemm1_k<<<g1, 256, 0, stream>>>(x, abf, gate, zg);

  dim3 g2(NTOK / BM, OUT_F / BN);    // (128, 32)
  gemm2_k<<<g2, 256, 0, stream>>>(zg, bbf, out);
}
